// MultiViewVideoBackboneV12_79611513798996
// MI455X (gfx1250) — matrix operand order repaired and checked by reading, not compiled, not run on hardware
//
#include <hip/hip_runtime.h>
#include <hip/hip_bf16.h>

// ---------------------------------------------------------------------------
// MultiViewVideoBackbone for MI455X (gfx1250, wave32, WMMA bf16, f32 accum)
//
// Implicit-GEMM convs through v_wmma_f32_16x16x32_bf16 with compile-time
// shapes, bf16 NHWC activations (channel-contiguous -> wide b128 loads),
// MT x NT register blocking so each B fragment feeds MT WMMAs and each A
// fragment feeds NT WMMAs.
// ---------------------------------------------------------------------------

typedef __attribute__((ext_vector_type(16))) __bf16 v16bf;
typedef __attribute__((ext_vector_type(8)))  __bf16 v8bf;
typedef __attribute__((ext_vector_type(8)))  float  v8f;

#define BN_S_CONST 0.9999950000374997f  /* 1/sqrt(1+1e-5) */

__device__ __forceinline__ float gelu_exact(float x) {
    return 0.5f * x * (1.0f + erff(x * 0.70710678118654752440f));
}

// ---------------------------------------------------------------------------
// Implicit-GEMM conv, one wave per (16*MT Cout) x (16*NT pixels) tile.
//   xT : bf16 NHWC  [img][H*W][CIN]
//   wT : bf16       [KS*KS][COUT][CIN]
//   OUTMODE: 0 = f32 NCHW, 1 = bf16 NHWC, 2 = f32 NHWC
//
// WMMA 16x16x32 bf16 operand layouts (CDNA5 ISA 7.12.2, wave32):
//   A: lane holds row M=lane&15; lanes<16 -> K 0-15, lanes>=16 -> K 16-31 (contiguous, as loaded here)
//   B: lane holds col N=lane&15; lanes<16 -> K 0-15, lanes>=16 -> K 16-31 (same order as A; not the ISA table order)
//   C/D: VGPR r -> M = 8*(lane>=16) + r, N = lane&15
// ---------------------------------------------------------------------------
template<int CIN, int COUT, int H, int W, int KS, int PAD,
         int OUTMODE, int DOGELU, int MT, int NT>
__global__ __launch_bounds__(32)
void conv_wmma(const __bf16* __restrict__ xT,
               const __bf16* __restrict__ wT,
               const float*  __restrict__ bias,
               void* __restrict__ yout)
{
    constexpr int HW = H * W;
    const int img = blockIdx.z;
    const int m0  = blockIdx.y * (16 * MT);
    const int n0  = blockIdx.x * (16 * NT);

    const int lane = threadIdx.x;
    const int half = lane >> 4;
    const int l15  = lane & 15;
    const int aK   = half * 16;   // A: K base
    const int bK   = half * 16;   // B: K base

    int nn[NT], oy[NT], ox[NT];
    #pragma unroll
    for (int t = 0; t < NT; ++t) {
        nn[t] = n0 + t * 16 + l15;
        oy[t] = nn[t] / W;                 // W compile-time: strength-reduced
        ox[t] = nn[t] - oy[t] * W;
    }

    const v8f zero8 = {0.f, 0.f, 0.f, 0.f, 0.f, 0.f, 0.f, 0.f};
    v8f acc[MT][NT];
    #pragma unroll
    for (int mt = 0; mt < MT; ++mt)
        #pragma unroll
        for (int t = 0; t < NT; ++t) acc[mt][t] = zero8;

    const __bf16* xbase = xT + (size_t)img * HW * CIN;

    #pragma unroll
    for (int ky = 0; ky < KS; ++ky) {
        #pragma unroll
        for (int kx = 0; kx < KS; ++kx) {
            const int tap = ky * KS + kx;
            const __bf16* wtap = wT + (size_t)tap * COUT * CIN;
            __builtin_prefetch(wtap + (size_t)(m0 + l15) * CIN, 0, 1);
            #pragma unroll
            for (int c0 = 0; c0 < CIN; c0 += 32) {
                // A fragments: MT x one contiguous 32B load
                v16bf a[MT];
                #pragma unroll
                for (int mt = 0; mt < MT; ++mt) {
                    const __bf16* wr =
                        wtap + (size_t)(m0 + mt * 16 + l15) * CIN + c0;
                    a[mt] = *(const v16bf*)(wr + aK);
                }
                #pragma unroll
                for (int t = 0; t < NT; ++t) {
                    const int py = oy[t] + ky - PAD;
                    const int px = ox[t] + kx - PAD;
                    const bool ok = (py >= 0) && (py < H) &&
                                    (px >= 0) && (px < W);
                    v16bf b = {};
                    if (ok)
                        b = *(const v16bf*)(xbase +
                              ((size_t)(py * W + px)) * CIN + c0 + bK);
                    #pragma unroll
                    for (int mt = 0; mt < MT; ++mt)
                        acc[mt][t] = __builtin_amdgcn_wmma_f32_16x16x32_bf16(
                                       false, a[mt], false, b, (short)0,
                                       acc[mt][t], false, false);
                }
            }
        }
    }

    #pragma unroll
    for (int mt = 0; mt < MT; ++mt) {
        const int mb = m0 + mt * 16 + half * 8;
        const v8f bv = *(const v8f*)(bias + mb);   // 32B vector bias load
        #pragma unroll
        for (int t = 0; t < NT; ++t) {
            float vals[8];
            #pragma unroll
            for (int r = 0; r < 8; ++r) {
                float v = acc[mt][t][r] + bv[r];
                if (DOGELU) v = gelu_exact(v * BN_S_CONST);
                vals[r] = v;
            }
            const int n = nn[t];
            if (OUTMODE == 0) {                 // f32 NCHW (final output)
                float* y = (float*)yout + (size_t)img * COUT * HW;
                #pragma unroll
                for (int r = 0; r < 8; ++r)
                    y[(size_t)(mb + r) * HW + n] = vals[r];
            } else if (OUTMODE == 1) {          // bf16 NHWC: one 16B store
                __bf16* y = (__bf16*)yout +
                    ((size_t)img * HW + n) * COUT + mb;
                v8bf o;
                #pragma unroll
                for (int r = 0; r < 8; ++r) o[r] = (__bf16)vals[r];
                *(v8bf*)y = o;
            } else {                            // f32 NHWC: one 32B store
                float* y = (float*)yout +
                    ((size_t)img * HW + n) * COUT + mb;
                v8f o;
                #pragma unroll
                for (int r = 0; r < 8; ++r) o[r] = vals[r];
                *(v8f*)y = o;
            }
        }
    }
}

// ---------------------------------------------------------------------------
// Packing kernels
// ---------------------------------------------------------------------------
template<int C, int HW>
__global__ __launch_bounds__(256)
void pack_nchw_to_nhwc_bf16(const float* __restrict__ x,
                            __bf16* __restrict__ xT, int nImg)
{
    const int t = blockIdx.x * 256 + threadIdx.x;
    if (t >= nImg * C * HW) return;
    const int c    = t % C;
    const int rest = t / C;
    const int n    = rest % HW;
    const int img  = rest / HW;
    xT[t] = (__bf16)x[((size_t)img * C + c) * HW + n];
}

template<int CIN, int COUT, int KS>
__global__ __launch_bounds__(256)
void pack_weights_bf16(const float* __restrict__ w, __bf16* __restrict__ wT)
{
    constexpr int KS2 = KS * KS;
    const int t = blockIdx.x * 256 + threadIdx.x;
    if (t >= KS2 * COUT * CIN) return;
    const int k    = t % CIN;
    const int rest = t / CIN;
    const int m    = rest % COUT;
    const int tap  = rest / COUT;
    wT[t] = (__bf16)w[((size_t)m * CIN + k) * KS2 + tap];
}

__global__ __launch_bounds__(256)
void f32_to_bf16_kernel(const float* __restrict__ x,
                        __bf16* __restrict__ y, int n)
{
    const int t = blockIdx.x * 256 + threadIdx.x;
    if (t < n) y[t] = (__bf16)x[t];
}

// ---------------------------------------------------------------------------
// Softmax over 32 depth bins; dl is f32 NHWC: [(img*1792+p)*32 + d]
// ---------------------------------------------------------------------------
__global__ __launch_bounds__(256)
void depth_softmax_nhwc(float* __restrict__ dl)
{
    const int NPIX = 12 * 1792;
    const int t = blockIdx.x * 256 + threadIdx.x;
    if (t >= NPIX) return;
    float* base = dl + (size_t)t * 32;
    float e[32];
    float mx = -3.4e38f;
    #pragma unroll
    for (int d = 0; d < 32; ++d) { e[d] = base[d]; mx = fmaxf(mx, e[d]); }
    float s = 0.f;
    #pragma unroll
    for (int d = 0; d < 32; ++d) { e[d] = expf(e[d] - mx); s += e[d]; }
    const float inv = 1.0f / s;
    #pragma unroll
    for (int d = 0; d < 32; ++d) base[d] = e[d] * inv;
}

// ---------------------------------------------------------------------------
// Lift-splat scatter, one thread per (img, pixel, depth).
//   fpT  f32 NHWC [(img*1792+p)*128 + cb]
//   prob f32 NHWC [(img*1792+p)*32 + d]
//   bev  f32 cell-major [(b*16384 + cell)*128 + cb] -> 128 consecutive atomics
// (int) casts truncate toward zero, matching .astype(int32).
// ---------------------------------------------------------------------------
__global__ __launch_bounds__(256)
void splat_scatter(const float* __restrict__ fpT, const float* __restrict__ prob,
                   const float* __restrict__ Kmat, const float* __restrict__ Tmat,
                   const float* __restrict__ trust, const float* __restrict__ depths,
                   float* __restrict__ bev)
{
    const int NP = 1792, NDb = 32, Vv = 6;
    const long total = 12L * NP * NDb;
    const long t = (long)blockIdx.x * blockDim.x + threadIdx.x;
    if (t >= total) return;

    const int img = (int)(t / (NP * NDb));
    const int rem = (int)(t - (long)img * NP * NDb);
    const int p   = rem / NDb;
    const int d   = rem - p * NDb;
    const int bi  = img / Vv;

    const float px = (float)(p % 56);
    const float py = (float)(p / 56);

    const float* Kk = Kmat + (size_t)img * 9;
    const float k00 = Kk[0], k01 = Kk[1], k02 = Kk[2];
    const float k10 = Kk[3], k11 = Kk[4], k12 = Kk[5];
    const float k20 = Kk[6], k21 = Kk[7], k22 = Kk[8];
    const float det = k00 * (k11 * k22 - k12 * k21)
                    - k01 * (k10 * k22 - k12 * k20)
                    + k02 * (k10 * k21 - k11 * k20);
    const float id = 1.0f / det;
    const float i00 =  (k11 * k22 - k12 * k21) * id;
    const float i01 = -(k01 * k22 - k02 * k21) * id;
    const float i02 =  (k01 * k12 - k02 * k11) * id;
    const float i10 = -(k10 * k22 - k12 * k20) * id;
    const float i11 =  (k00 * k22 - k02 * k20) * id;
    const float i12 = -(k00 * k12 - k02 * k10) * id;
    const float i20 =  (k10 * k21 - k11 * k20) * id;
    const float i21 = -(k00 * k21 - k01 * k20) * id;
    const float i22 =  (k00 * k11 - k01 * k10) * id;

    const float rx = i00 * px + i01 * py + i02;
    const float ry = i10 * px + i11 * py + i12;
    const float rz = i20 * px + i21 * py + i22;

    const float dep = depths[d];
    const float X = rx * dep, Y = ry * dep, Z = rz * dep;

    const float* Tt = Tmat + (size_t)img * 16;
    const float xe = Tt[0] * X + Tt[1] * Y + Tt[2] * Z + Tt[3];
    const float ye = Tt[4] * X + Tt[5] * Y + Tt[6] * Z + Tt[7];

    const float fc = (xe + 20.0f) / 40.0f * 127.0f;
    const float fr = (ye + 20.0f) / 40.0f * 127.0f;
    const int col = (int)fc;
    const int row = (int)fr;
    if (col < 0 || col >= 128 || row < 0 || row >= 128) return;

    const float scale = prob[((size_t)img * NP + p) * 32 + d] * trust[img];
    const float* fpp = fpT + ((size_t)img * NP + p) * 128;
    float* bb = bev + ((size_t)bi * 16384 + (row * 128 + col)) * 128;

    #pragma unroll 4
    for (int cb = 0; cb < 128; ++cb)
        atomicAdd(bb + cb, fpp[cb] * scale);
}

// ---------------------------------------------------------------------------
// Workspace layout (bytes), peak ~36.8 MB:
//   [0,        16777216)  bev   f32 cell-major      (dies after bf16 convert)
//   [16777216, 16850944)  dh_w1T bf16
//   [16850944, 16855040)  dh_w2T bf16
//   [16855040, 16871424)  fp_wT  bf16
//   [16871424, 17166336)  br_w1T bf16
//   [17166336, 17461248)  br_w2T bf16
//   [17461248, 20213760)  featT bf16 NHWC          (dies after fp conv)
//   [20213760, 22966272)  hT    bf16 NHWC          (dies after logits conv)
//   [22966272, 25718784)  dl    f32 NHWC           (dies after scatter)
//   [25718784, 36728832)  fpT   f32 NHWC           (dies after scatter)
//   bevT bf16 aliases 17461248 (over dead featT/hT/dl)
//   h2T  bf16 aliases 0        (over dead bev)
// ---------------------------------------------------------------------------
extern "C" void kernel_launch(void* const* d_in, const int* in_sizes, int n_in,
                              void* d_out, int out_size, void* d_ws, size_t ws_size,
                              hipStream_t stream)
{
    (void)in_sizes; (void)n_in; (void)out_size; (void)ws_size;

    const float* feat   = (const float*)d_in[0];
    const float* Kmat   = (const float*)d_in[1];
    const float* Tmat   = (const float*)d_in[2];
    const float* trust  = (const float*)d_in[3];
    const float* depths = (const float*)d_in[4];
    const float* dh_w1  = (const float*)d_in[5];
    const float* dh_b1  = (const float*)d_in[6];
    const float* dh_w2  = (const float*)d_in[7];
    const float* dh_b2  = (const float*)d_in[8];
    const float* fp_w   = (const float*)d_in[9];
    const float* fp_b   = (const float*)d_in[10];
    const float* br_w1  = (const float*)d_in[11];
    const float* br_b1  = (const float*)d_in[12];
    const float* br_w2  = (const float*)d_in[13];
    const float* br_b2  = (const float*)d_in[14];
    float* out = (float*)d_out;

    char* ws = (char*)d_ws;
    float*  bev     = (float*) (ws);
    __bf16* dh_w1T  = (__bf16*)(ws + 16777216);
    __bf16* dh_w2T  = (__bf16*)(ws + 16850944);
    __bf16* fp_wT   = (__bf16*)(ws + 16855040);
    __bf16* br_w1T  = (__bf16*)(ws + 16871424);
    __bf16* br_w2T  = (__bf16*)(ws + 17166336);
    __bf16* featT   = (__bf16*)(ws + 17461248);
    __bf16* hT      = (__bf16*)(ws + 20213760);
    float*  dl      = (float*) (ws + 22966272);
    float*  fpT     = (float*) (ws + 25718784);
    __bf16* bevT    = (__bf16*)(ws + 17461248);   // alias
    __bf16* h2T     = (__bf16*)(ws);              // alias

    hipMemsetAsync(bev, 0, (size_t)2 * 16384 * 128 * sizeof(float), stream);

    // ---- pack weights + activations to bf16 -------------------------------
    pack_weights_bf16<64, 64, 3><<<(9*64*64 + 255)/256, 256, 0, stream>>>(dh_w1, dh_w1T);
    pack_weights_bf16<64, 32, 1><<<(32*64   + 255)/256, 256, 0, stream>>>(dh_w2, dh_w2T);
    pack_weights_bf16<64, 128,1><<<(128*64  + 255)/256, 256, 0, stream>>>(fp_w,  fp_wT);
    pack_weights_bf16<128,128,3><<<(9*128*128+255)/256, 256, 0, stream>>>(br_w1, br_w1T);
    pack_weights_bf16<128,128,3><<<(9*128*128+255)/256, 256, 0, stream>>>(br_w2, br_w2T);
    pack_nchw_to_nhwc_bf16<64, 1792><<<(12*64*1792 + 255)/256, 256, 0, stream>>>(feat, featT, 12);

    const dim3 wblk(32, 1, 1);

    // 1) depth-head conv1: 3x3 64->64, GELU -> hT (bf16 NHWC)
    conv_wmma<64, 64, 32, 56, 3, 1, /*OUT*/1, /*GELU*/1, /*MT*/2, /*NT*/4>
        <<<dim3(28, 2, 12), wblk, 0, stream>>>(featT, dh_w1T, dh_b1, hT);

    // 2) depth logits: 1x1 64->32 -> dl (f32 NHWC)
    conv_wmma<64, 32, 32, 56, 1, 0, /*OUT*/2, /*GELU*/0, /*MT*/2, /*NT*/4>
        <<<dim3(28, 1, 12), wblk, 0, stream>>>(hT, dh_w2T, dh_b2, dl);

    // 3) softmax over depth bins (contiguous 32 floats/pixel)
    depth_softmax_nhwc<<<(12*1792 + 255)/256, 256, 0, stream>>>(dl);

    // 4) feature projection: 1x1 64->128, GELU -> fpT (f32 NHWC)
    conv_wmma<64, 128, 32, 56, 1, 0, /*OUT*/2, /*GELU*/1, /*MT*/2, /*NT*/4>
        <<<dim3(28, 4, 12), wblk, 0, stream>>>(featT, fp_wT, fp_b, fpT);

    // 5) lift-splat scatter into cell-major BEV
    splat_scatter<<<(12*1792*32 + 255)/256, 256, 0, stream>>>(
        fpT, dl, Kmat, Tmat, trust, depths, bev);

    // 5b) BEV f32 -> bf16 (same cell-major layout)
    f32_to_bf16_kernel<<<(2*16384*128 + 255)/256, 256, 0, stream>>>(
        bev, bevT, 2*16384*128);

    // 6) BEV refine conv1: 3x3 128->128, GELU -> h2T (bf16 NHWC)
    conv_wmma<128, 128, 128, 128, 3, 1, /*OUT*/1, /*GELU*/1, /*MT*/2, /*NT*/4>
        <<<dim3(256, 4, 2), wblk, 0, stream>>>(bevT, br_w1T, br_b1, h2T);

    // 7) BEV refine conv2: 3x3 128->128, GELU -> out (f32 NCHW)
    conv_wmma<128, 128, 128, 128, 3, 1, /*OUT*/0, /*GELU*/1, /*MT*/2, /*NT*/4>
        <<<dim3(256, 4, 2), wblk, 0, stream>>>(h2T, br_w2T, br_b2, out);
}
